// EncoderDecoder_9259949490916
// MI455X (gfx1250) — compile-verified, matrix operand order checked by reading, not run on hardware
//
#include <hip/hip_runtime.h>
#include <hip/hip_bf16.h>
#include <stdint.h>
#include <stddef.h>

// ---------------------------------------------------------------------------
// Problem constants (match reference)
// ---------------------------------------------------------------------------
#define B_DIM 64
#define T_DIM 32
#define C_DIM 2048
#define D_DIM 1024
#define F_IN   1.0e-4f
#define F_LAT  1.0e-3f
#define LN_EPS 1.0e-5f

typedef __attribute__((ext_vector_type(16))) __bf16 v16bf;
typedef __attribute__((ext_vector_type(2)))  __bf16 v2bf;
typedef __attribute__((ext_vector_type(8)))  float  v8f;
typedef __attribute__((ext_vector_type(4)))  float  f32x4;
typedef __attribute__((ext_vector_type(4)))  int    i32x4;

// ---------------------------------------------------------------------------
// fp32 -> bf16 helpers: native fptrunc (RNE) -> v_cvt_pk_bf16_f32 on gfx1250
// ---------------------------------------------------------------------------
__device__ __forceinline__ uint16_t f2bf_bits(float f) {
    __bf16 b = (__bf16)f;
    uint16_t u; __builtin_memcpy(&u, &b, 2);
    return u;
}

__device__ __forceinline__ uint32_t pack2bf(float a, float b) {
    v2bf v = { (__bf16)a, (__bf16)b };
    uint32_t u; __builtin_memcpy(&u, &v, 4);
    return u;
}

__device__ __forceinline__ float gelu_exact(float v) {
    return 0.5f * v * (1.0f + erff(v * 0.70710678118654752f));
}

// ---------------------------------------------------------------------------
// Async global->LDS copy (CDNA5 GLOBAL_LOAD_ASYNC_TO_LDS_B128, ASYNCcnt)
// clang-22 signature: (int4* gsrc, int4* ldst, imm offset, imm cpol)
// ---------------------------------------------------------------------------
#if __has_builtin(__builtin_amdgcn_global_load_async_to_lds_b128)
#define HAVE_ASYNC_LDS 1
#else
#define HAVE_ASYNC_LDS 0
#endif

__device__ __forceinline__ void copy16_g2lds(const void* g, void* l) {
#if HAVE_ASYNC_LDS
    __builtin_amdgcn_global_load_async_to_lds_b128(
        (i32x4*)const_cast<void*>(g), (i32x4*)l, 0, 0);
#else
    *(uint4*)l = *(const uint4*)g;
#endif
}

__device__ __forceinline__ void async_wait_all() {
#if HAVE_ASYNC_LDS
#if __has_builtin(__builtin_amdgcn_s_wait_asynccnt)
    __builtin_amdgcn_s_wait_asynccnt(0);
#else
    asm volatile("s_wait_asynccnt 0" ::: "memory");
#endif
#endif
}

// Assemble a v16bf WMMA fragment from an LDS row laid out with K contiguous.
// 16-bit operand layout (ISA 7.12.2): element i of lane (m, hi=lane>>4) holds
// k = 16*(i/8) + 8*hi + (i%8)  ->  two contiguous 8-element (16B) chunks.
__device__ __forceinline__ v16bf frag_from_lds(const uint16_t* row, int hi) {
    v16bf f;
    __builtin_memcpy(&f,                               row + 8 * hi,      16);
    __builtin_memcpy(reinterpret_cast<char*>(&f) + 16, row + 16 + 8 * hi, 16);
    return f;
}

// ---------------------------------------------------------------------------
// Generic batched-GEMM tile kernel.
//   D[b, n] = epilogue( sum_k A[b, k] * W[t, k, n] )    with M = B_DIM = 64
// grid = (NTOT/128, T).  block = 256 (8 wave32s).  64x128 output tile / block.
// K-tile = 64: 24 KB LDS, 8 WMMAs per wave per stage (4 acc tiles / wave).
// K, NTOT, ORS are compile-time so all W loads / LDS stores / output stores
// fold into base + imm24 offsets (no per-iteration 64-bit address chains).
// A_F32    : A read from fp32 x with input-noise mixing; else A is bf16 ws.
// EPI_GELU : apply exact GELU before store.
// EPI_BF16 : store bf16 to Ob, else fp32 to Of.
// ---------------------------------------------------------------------------
template<int K, int NTOT, int ORS, bool A_F32, bool EPI_GELU, bool EPI_BF16>
__launch_bounds__(256)
__global__ void gemm64_wmma(const float* __restrict__ Af, const float* __restrict__ An,
                            const uint16_t* __restrict__ Ab,
                            const float* __restrict__ W,
                            float* __restrict__ Of, uint16_t* __restrict__ Ob,
                            int a_row_stride, int a_t_mul, int o_t_mul) {
    __shared__ alignas(16) uint16_t As[64 * 64];    // [m][k] bf16, k contiguous
    __shared__ alignas(16) uint16_t Bs[128 * 64];   // [n][k] bf16 (K-transposed)

    const int tid  = threadIdx.x;
    const int lane = tid & 31;
    const int wv   = tid >> 5;               // 0..7
    const int mi   = wv & 3;                 // m-tile (16 rows each)
    const int ng   = (wv >> 2) * 4;          // first of four n-tiles
    const int l15  = lane & 15;
    const int hi   = lane >> 4;

    const int t  = blockIdx.y;
    const int n0 = blockIdx.x * 128;
    const size_t a_off = (size_t)t * a_t_mul;
    const size_t o_off = (size_t)t * o_t_mul;

    // ---- per-thread staging bases (hoisted; inner offsets are constants) ---
    // A tile: thread covers row sm, 16 k's starting at sk
    const int sm = tid >> 2;
    const int sk = (tid & 3) * 16;
    const size_t a_base = (size_t)sm * a_row_stride + a_off + sk;
    uint16_t* const As_dst = &As[sm * 64 + sk];

    // W tile: thread covers k rows {kb + 16*it, +1} x 4 n's at nb
    const int kb = 2 * (tid >> 5);
    const int nb = 4 * (tid & 31);
    const float* __restrict__ wp =
        W + (size_t)t * K * NTOT + (size_t)kb * NTOT + n0 + nb;
    uint16_t* const bs_dst = &Bs[nb * 64 + kb];

    v8f acc[4];
    acc[0] = (v8f){}; acc[1] = (v8f){}; acc[2] = (v8f){}; acc[3] = (v8f){};

    for (int k0 = 0; k0 < K; k0 += 64) {
        // ---- stage A tile: 64x64 bf16, 16 elements (32B) per thread --------
        if (A_F32) {
            uint32_t pk[8];
            #pragma unroll
            for (int c = 0; c < 4; ++c) {
                const f32x4 xv = *(const f32x4*)&Af[a_base + k0 + c * 4];
                const f32x4 nv = *(const f32x4*)&An[a_base + k0 + c * 4];
                const float f0 = nv.x * F_IN + xv.x * (1.0f - F_IN);
                const float f1 = nv.y * F_IN + xv.y * (1.0f - F_IN);
                const float f2 = nv.z * F_IN + xv.z * (1.0f - F_IN);
                const float f3 = nv.w * F_IN + xv.w * (1.0f - F_IN);
                pk[c * 2]     = pack2bf(f0, f1);
                pk[c * 2 + 1] = pack2bf(f2, f3);
            }
            *(uint4*)(As_dst)     = *(const uint4*)&pk[0];
            *(uint4*)(As_dst + 8) = *(const uint4*)&pk[4];
        } else {
            // pure byte copy: use CDNA5 async global->LDS DMA (ASYNCcnt)
            const uint16_t* g = Ab + a_base + k0;
            copy16_g2lds(g,     As_dst);
            copy16_g2lds(g + 8, As_dst + 8);
        }

        // ---- stage W tile transposed: 64x128 -> Bs[n][k] -------------------
        // weights are single-use: non-temporal loads keep them out of L2's way
        #pragma unroll
        for (int it = 0; it < 4; ++it) {
            const float* r0p = wp + (size_t)(k0 + it * 16) * NTOT;
            const f32x4 w0 = __builtin_nontemporal_load((const f32x4*)r0p);
            const f32x4 w1 = __builtin_nontemporal_load((const f32x4*)(r0p + NTOT));
            *(uint32_t*)&bs_dst[0 * 64 + it * 16] = pack2bf(w0.x, w1.x);
            *(uint32_t*)&bs_dst[1 * 64 + it * 16] = pack2bf(w0.y, w1.y);
            *(uint32_t*)&bs_dst[2 * 64 + it * 16] = pack2bf(w0.z, w1.z);
            *(uint32_t*)&bs_dst[3 * 64 + it * 16] = pack2bf(w0.w, w1.w);
        }

        // prefetch next weight K-tile while we compute (global_prefetch_b8)
        if (k0 + 64 < K) {
            __builtin_prefetch(wp + (size_t)(k0 + 64) * NTOT, 0, 1);
        }

        async_wait_all();
        __syncthreads();

        // ---- 8 WMMAs per wave: per K-half, preload all fragments then chain
        #pragma unroll
        for (int kk = 0; kk < 64; kk += 32) {
            const v16bf afrag = frag_from_lds(&As[(mi * 16 + l15) * 64 + kk], hi);
            v16bf bfrag[4];
            #pragma unroll
            for (int j = 0; j < 4; ++j)
                bfrag[j] = frag_from_lds(&Bs[((ng + j) * 16 + l15) * 64 + kk], hi);
            #pragma unroll
            for (int j = 0; j < 4; ++j)
                acc[j] = __builtin_amdgcn_wmma_f32_16x16x32_bf16(
                             false, afrag, false, bfrag[j], (short)0, acc[j], false, false);
        }
        __syncthreads();
    }

    // ---- epilogue: C/D layout -> lane(l15)=n, VGPR r -> m = 16*mi + 8*hi + r
    const int m0 = mi * 16 + hi * 8;
    #pragma unroll
    for (int j = 0; j < 4; ++j) {
        const int n = n0 + (ng + j) * 16 + l15;
        if (EPI_BF16) {
            uint16_t* op = Ob + (size_t)m0 * ORS + o_off + n;
            #pragma unroll
            for (int r = 0; r < 8; ++r) {
                float v = acc[j][r];
                if (EPI_GELU) v = gelu_exact(v);
                op[(size_t)r * ORS] = f2bf_bits(v);
            }
        } else {
            float* op = Of + (size_t)m0 * ORS + o_off + n;
            #pragma unroll
            for (int r = 0; r < 8; ++r) {
                float v = acc[j][r];
                if (EPI_GELU) v = gelu_exact(v);
                op[(size_t)r * ORS] = v;
            }
        }
    }
}

// ---------------------------------------------------------------------------
// LayerNorm + latent noise + clamp:  Z = clip(noise_z*F_LAT + LN(H2)*(1-F_LAT))
// one block per row (b*T + t), 256 threads, D = 1024 (4 elems/thread, f32x4)
// ---------------------------------------------------------------------------
__launch_bounds__(256)
__global__ void ln_noise_clamp(const float* __restrict__ H2,
                               const float* __restrict__ noise_z,
                               const float* __restrict__ gamma,
                               const float* __restrict__ beta,
                               uint16_t* __restrict__ Z) {
    const int row = blockIdx.x;                    // b*T + t
    const int tid = threadIdx.x;
    const size_t base = (size_t)row * D_DIM + tid * 4;

    const f32x4 h = *(const f32x4*)(H2 + base);
    float s  = h.x + h.y + h.z + h.w;
    float ss = h.x * h.x + h.y * h.y + h.z * h.z + h.w * h.w;

    __shared__ float r0[256], r1[256];
    r0[tid] = s; r1[tid] = ss;
    __syncthreads();
    for (int w = 128; w >= 1; w >>= 1) {
        if (tid < w) { r0[tid] += r0[tid + w]; r1[tid] += r1[tid + w]; }
        __syncthreads();
    }
    const float mu  = r0[0] * (1.0f / D_DIM);
    const float var = r1[0] * (1.0f / D_DIM) - mu * mu;
    const float inv = rsqrtf(var + LN_EPS);

    const f32x4 nz = *(const f32x4*)(noise_z + base);
    const f32x4 gm = *(const f32x4*)(gamma + tid * 4);
    const f32x4 bt = *(const f32x4*)(beta  + tid * 4);

    float z[4];
    z[0] = (h.x - mu) * inv * gm.x + bt.x;
    z[1] = (h.y - mu) * inv * gm.y + bt.y;
    z[2] = (h.z - mu) * inv * gm.z + bt.z;
    z[3] = (h.w - mu) * inv * gm.w + bt.w;
    z[0] = fminf(1.f, fmaxf(-1.f, nz.x * F_LAT + z[0] * (1.0f - F_LAT)));
    z[1] = fminf(1.f, fmaxf(-1.f, nz.y * F_LAT + z[1] * (1.0f - F_LAT)));
    z[2] = fminf(1.f, fmaxf(-1.f, nz.z * F_LAT + z[2] * (1.0f - F_LAT)));
    z[3] = fminf(1.f, fmaxf(-1.f, nz.w * F_LAT + z[3] * (1.0f - F_LAT)));

    uint2 packed;
    packed.x = pack2bf(z[0], z[1]);
    packed.y = pack2bf(z[2], z[3]);
    *(uint2*)(Z + base) = packed;
}

// ---------------------------------------------------------------------------
// launch
// ---------------------------------------------------------------------------
extern "C" void kernel_launch(void* const* d_in, const int* in_sizes, int n_in,
                              void* d_out, int out_size, void* d_ws, size_t ws_size,
                              hipStream_t stream) {
    (void)in_sizes; (void)n_in; (void)out_size; (void)ws_size;
    const float* x       = (const float*)d_in[0];
    const float* noise_x = (const float*)d_in[1];
    const float* noise_z = (const float*)d_in[2];
    const float* enc_w1  = (const float*)d_in[3];
    const float* enc_w2  = (const float*)d_in[4];
    const float* dec_w1  = (const float*)d_in[5];
    const float* dec_w2  = (const float*)d_in[6];
    const float* gamma   = (const float*)d_in[7];
    const float* beta    = (const float*)d_in[8];
    float* out = (float*)d_out;

    // workspace partition
    char* ws = (char*)d_ws;
    uint16_t* H1 = (uint16_t*)(ws);                          // 64*32*1024 bf16 = 4 MiB
    float*    H2 = (float*)   (ws + (size_t)(4  << 20));     // 64*32*1024 f32  = 8 MiB
    uint16_t* Z  = (uint16_t*)(ws + (size_t)(12 << 20));     // 4 MiB
    uint16_t* G  = (uint16_t*)(ws + (size_t)(16 << 20));     // 4 MiB

    dim3 blk(256);

    // enc1: H1 = gelu( (noise_x*F_IN + x*(1-F_IN)) @ enc_w1[t] )   K=2048, N=1024
    gemm64_wmma<C_DIM, D_DIM, T_DIM * D_DIM, true, true, true>
        <<<dim3(D_DIM / 128, T_DIM), blk, 0, stream>>>(
        x, noise_x, nullptr, enc_w1, nullptr, H1,
        T_DIM * C_DIM, C_DIM, D_DIM);

    // enc2: H2 = H1 @ enc_w2[t]                                   K=1024, N=1024
    gemm64_wmma<D_DIM, D_DIM, T_DIM * D_DIM, false, false, false>
        <<<dim3(D_DIM / 128, T_DIM), blk, 0, stream>>>(
        nullptr, nullptr, H1, enc_w2, H2, nullptr,
        T_DIM * D_DIM, D_DIM, D_DIM);

    // LN + latent noise + clamp -> Z (bf16)
    ln_noise_clamp<<<dim3(B_DIM * T_DIM), blk, 0, stream>>>(H2, noise_z, gamma, beta, Z);

    // dec1: G = gelu( Z @ dec_w1[t] )                             K=1024, N=1024
    gemm64_wmma<D_DIM, D_DIM, T_DIM * D_DIM, false, true, true>
        <<<dim3(D_DIM / 128, T_DIM), blk, 0, stream>>>(
        nullptr, nullptr, Z, dec_w1, nullptr, G,
        T_DIM * D_DIM, D_DIM, D_DIM);

    // dec2: out = G @ dec_w2[t]                                   K=1024, N=2048
    gemm64_wmma<D_DIM, C_DIM, T_DIM * C_DIM, false, false, false>
        <<<dim3(C_DIM / 128, T_DIM), blk, 0, stream>>>(
        nullptr, nullptr, G, dec_w2, out, nullptr,
        T_DIM * D_DIM, D_DIM, C_DIM);
}
